// BottleneckAttention_5729486373612
// MI455X (gfx1250) — matrix operand order repaired and checked by reading, not compiled, not run on hardware
//
#include <hip/hip_runtime.h>

// ---------------------------------------------------------------------------
// BottleneckAttention on gfx1250 (MI455X).
// bf16 WMMA (v_wmma_f32_16x16x32_bf16) for all three GEMM stages, fp32
// accumulate, fp32 softmax + rel-pos bias.  Roofline: ~125 GFLOP / ~0.5 GB of
// traffic -> bf16 matrix path keeps compute (~25us) balanced with HBM (~22us);
// the f32 WMMA (16x16x4) path would be ~10x compute-bound.
// Workspace: ~203 MB (xT, wbf, rel tables, Q, K(scaled), V^T, Pw, Ph).
// ---------------------------------------------------------------------------

typedef unsigned short u16;
typedef __attribute__((ext_vector_type(8)))  u16    v8u;
typedef __attribute__((ext_vector_type(16))) u16    v16u;
typedef __attribute__((ext_vector_type(8)))  __bf16 v8bf;
typedef __attribute__((ext_vector_type(16))) __bf16 v16bf;
typedef __attribute__((ext_vector_type(8)))  float  v8f;

#define NB    32            // batch
#define NH    4             // heads
#define DH    128           // dim_head
#define CIN   512           // input channels
#define OC    1536          // 3*heads*dim_head
#define HW    32            // H == W
#define NN    1024          // N = H*W
#define SCALE 0.08838834764831845f   // 128^-0.5
#define NEG_INF (-3.402823466e38f)

static __device__ __forceinline__ u16 f2bf(float f) {
  unsigned u = __builtin_bit_cast(unsigned, f);
  u += 0x7FFFu + ((u >> 16) & 1u);          // round-to-nearest-even
  return (u16)(u >> 16);
}

static __device__ __forceinline__ v8f zero8() {
  v8f z;
#pragma unroll
  for (int i = 0; i < 8; ++i) z[i] = 0.0f;
  return z;
}

// A-fragment (16x32 bf16): lane holds M=lane&15; 16 contiguous K at c1 = kc*32 + 16*(lane>>4), p1 = p0+8
static __device__ __forceinline__ v16bf ld_frag2(const u16* p0, const u16* p1) {
  v8u a = *(const v8u*)p0;
  v8u b = *(const v8u*)p1;
  v16u r;
#pragma unroll
  for (int i = 0; i < 8; ++i) { r[i] = a[i]; r[i + 8] = b[i]; }
  return __builtin_bit_cast(v16bf, r);
}
// B-fragment (32x16 bf16): lane holds N=lane&15, 16 contiguous K at 16*(lane>>4)
static __device__ __forceinline__ v16bf ld_fragB(const u16* p) {
  return __builtin_bit_cast(v16bf, *(const v16u*)p);
}

static __device__ __forceinline__ v8f wmma_bf16(v16bf a, v16bf b, v8f c) {
  return __builtin_amdgcn_wmma_f32_16x16x32_bf16(false, a, false, b, (short)0, c,
                                                 false, false);
}

// ---------------------------------------------------------------- kernel 1 --
__global__ __launch_bounds__(256) void k_cvt_w(const float* __restrict__ w,
                                               u16* __restrict__ wbf) {
  int i = blockIdx.x * 256 + threadIdx.x;   // OC*CIN = 786432 elems
  wbf[i] = f2bf(w[i]);
}

// ---------------------------------------------------------------- kernel 2 --
// rel tables (63,128) f32 -> padded (64,128) bf16, row 63 zeroed.
__global__ __launch_bounds__(128) void k_cvt_rel(const float* __restrict__ relw,
                                                 const float* __restrict__ relh,
                                                 u16* __restrict__ relwbf,
                                                 u16* __restrict__ relhbf) {
  int t = blockIdx.x >> 6;
  int r = blockIdx.x & 63;
  int d = threadIdx.x;
  const float* src = t ? relh : relw;
  u16* dst = t ? relhbf : relwbf;
  float v = (r < 63) ? src[r * DH + d] : 0.0f;
  dst[r * DH + d] = f2bf(v);
}

// ---------------------------------------------------------------- kernel 3 --
// x (B,512,1024) f32  ->  xT (B,1024,512) bf16 via LDS transpose.
__global__ __launch_bounds__(256) void k_transpose_x(const float* __restrict__ x,
                                                     u16* __restrict__ xT) {
  __shared__ float tile[64][65];
  int b  = blockIdx.x >> 7;
  int rm = blockIdx.x & 127;
  int ct = (rm >> 4) * 64;
  int nt = (rm & 15) * 64;
#pragma unroll
  for (int it = 0; it < 16; ++it) {
    int idx = threadIdx.x + it * 256;
    int cc = idx >> 6, nn = idx & 63;
    tile[cc][nn] = x[((size_t)b * CIN + ct + cc) * NN + nt + nn];
  }
  __syncthreads();
#pragma unroll
  for (int it = 0; it < 16; ++it) {
    int idx = threadIdx.x + it * 256;
    int nn = idx >> 6, cc = idx & 63;
    xT[((size_t)b * NN + nt + nn) * CIN + ct + cc] = f2bf(tile[cc][nn]);
  }
}

// ---------------------------------------------------------------- kernel 4 --
// QKV GEMM: D[n,o] = sum_c xT[b,n,c] * w[o,c].  Each wave: 16(n) x 64(o).
// Epilogue decodes o = d*12 + k*4 + h -> Q(b,h,n,d), K(b,h,n,d)*scale, Vt(b,h,d,n).
__global__ __launch_bounds__(128) void k_qkv(const u16* __restrict__ xT,
                                             const u16* __restrict__ wbf,
                                             u16* __restrict__ Qb,
                                             u16* __restrict__ Kb,
                                             u16* __restrict__ Vt) {
  int wave = threadIdx.x >> 5, lane = threadIdx.x & 31;
  int nl = lane & 15, half = lane >> 4;
  int wt = blockIdx.x * 4 + wave;          // 49152 wave tiles
  int ot = wt % 24;
  int rest = wt / 24;
  int ntile = rest & 63;
  int b = rest >> 6;
  int n_base = ntile * 16, o_base = ot * 64;

  const u16* arow = xT + ((size_t)b * NN + n_base + nl) * CIN;
  v8f acc[4];
#pragma unroll
  for (int i = 0; i < 4; ++i) acc[i] = zero8();

#pragma unroll 4
  for (int kc = 0; kc < 16; ++kc) {
    int c1 = kc * 32 + 16 * half;
    v16bf A = ld_frag2(arow + c1, arow + c1 + 8);
#pragma unroll
    for (int os = 0; os < 4; ++os) {
      const u16* bp = wbf + (size_t)(o_base + os * 16 + nl) * CIN + kc * 32 + 16 * half;
      acc[os] = wmma_bf16(A, ld_fragB(bp), acc[os]);
    }
  }

  int nb = n_base + 8 * half;
#pragma unroll
  for (int os = 0; os < 4; ++os) {
    int o = o_base + os * 16 + nl;
    int d = o / 12;
    int rem = o - d * 12;
    int kk = rem >> 2, hh = rem & 3;
    size_t bh = (size_t)(b * NH + hh);
    if (kk == 2) {                          // V -> transposed (b,h,d,n), vector store
      v8u pv;
#pragma unroll
      for (int j = 0; j < 8; ++j) pv[j] = f2bf(acc[os][j]);
      *(v8u*)(Vt + (bh * DH + d) * NN + nb) = pv;
    } else {
      u16* dst = kk ? Kb : Qb;
      float s = kk ? SCALE : 1.0f;          // fold softmax scale into K
#pragma unroll
      for (int j = 0; j < 8; ++j)
        dst[(bh * NN + nb + j) * DH + d] = f2bf(acc[os][j] * s);
    }
  }
}

// ---------------------------------------------------------------- kernel 5 --
// Pw/Ph[b,h,i,r] = q[b,h,i,:] . rel[r,:]   (fp32 out, r padded to 64)
__global__ __launch_bounds__(128) void k_relpos(const u16* __restrict__ Qb,
                                                const u16* __restrict__ relwbf,
                                                const u16* __restrict__ relhbf,
                                                float* __restrict__ Pw,
                                                float* __restrict__ Ph) {
  int wave = threadIdx.x >> 5, lane = threadIdx.x & 31;
  int nl = lane & 15, half = lane >> 4;
  int wt = blockIdx.x * 4 + wave;          // 8192 wave tiles
  int itile = wt & 63;
  int bh = wt >> 6;
  int i_base = itile * 16;

  const u16* qrow = Qb + ((size_t)bh * NN + i_base + nl) * DH;
  v16bf A[4];
#pragma unroll
  for (int kc = 0; kc < 4; ++kc) {
    int c1 = kc * 32 + 16 * half;
    A[kc] = ld_frag2(qrow + c1, qrow + c1 + 8);
  }
#pragma unroll
  for (int t = 0; t < 2; ++t) {
    const u16* rb = t ? relhbf : relwbf;
    float* P = t ? Ph : Pw;
    v8f acc[4];
#pragma unroll
    for (int i = 0; i < 4; ++i) acc[i] = zero8();
#pragma unroll
    for (int kc = 0; kc < 4; ++kc)
#pragma unroll
      for (int rt = 0; rt < 4; ++rt) {
        const u16* bp = rb + (size_t)(rt * 16 + nl) * DH + kc * 32 + 16 * half;
        acc[rt] = wmma_bf16(A[kc], ld_fragB(bp), acc[rt]);
      }
#pragma unroll
    for (int rt = 0; rt < 4; ++rt) {
      int r = rt * 16 + nl;
#pragma unroll
      for (int j = 0; j < 8; ++j)
        P[((size_t)bh * NN + i_base + 8 * half + j) * 64 + r] = acc[rt][j];
    }
  }
}

// ---------------------------------------------------------------- kernel 6 --
// Fused flash attention with rel-pos bias. 4 waves/WG; wave w owns 16 query
// rows of a 64-row tile of one (b,h). Online softmax across 16 key blocks.
__global__ __launch_bounds__(128) void k_attn(const u16* __restrict__ Qb,
                                              const u16* __restrict__ Kb,
                                              const u16* __restrict__ Vt,
                                              const float* __restrict__ Pw,
                                              const float* __restrict__ Ph,
                                              float* __restrict__ out) {
  __shared__ float Pwl[64][64];
  __shared__ float Phl[64][64];
  __shared__ u16 Pbuf[4][16][64];          // per-wave C->A transpose buffer

  int wave = threadIdx.x >> 5, lane = threadIdx.x & 31;
  int nl = lane & 15, half = lane >> 4;
  int bh = blockIdx.x >> 4;
  int i0 = (blockIdx.x & 15) * 64;

  // stage rel-pos dot products for this WG's 64 query rows
  for (int idx = threadIdx.x; idx < 64 * 64; idx += 128) {
    int r = idx >> 6, c = idx & 63;
    size_t g = ((size_t)bh * NN + i0 + r) * 64 + c;
    Pwl[r][c] = Pw[g];
    Phl[r][c] = Ph[g];
  }
  __syncthreads();

  // Q fragments (16 rows x 128) held in registers
  const u16* qrow = Qb + ((size_t)bh * NN + i0 + wave * 16 + nl) * DH;
  v16bf QA[4];
#pragma unroll
  for (int kc = 0; kc < 4; ++kc) {
    int c1 = kc * 32 + 16 * half;
    QA[kc] = ld_frag2(qrow + c1, qrow + c1 + 8);
  }

  v8f O[8];
#pragma unroll
  for (int dt = 0; dt < 8; ++dt) O[dt] = zero8();
  float mrow[8], lrow[8];
#pragma unroll
  for (int j = 0; j < 8; ++j) { mrow[j] = NEG_INF; lrow[j] = 0.0f; }

  for (int jb = 0; jb < 16; ++jb) {
    int j0 = jb * 64;
    // ---- S = Q.K^T (K pre-scaled) ----
    v8f S[4];
#pragma unroll
    for (int jt = 0; jt < 4; ++jt) {
      v8f acc = zero8();
#pragma unroll
      for (int kc = 0; kc < 4; ++kc) {
        const u16* kp = Kb + ((size_t)bh * NN + j0 + jt * 16 + nl) * DH + kc * 32 + 16 * half;
        acc = wmma_bf16(QA[kc], ld_fragB(kp), acc);
      }
      S[jt] = acc;
    }
    // ---- add rel-pos bias ----
#pragma unroll
    for (int jt = 0; jt < 4; ++jt) {
      int jg = j0 + jt * 16 + nl;
      int yk = jg & 31, xk = jg >> 5;
#pragma unroll
      for (int jj = 0; jj < 8; ++jj) {
        int il = wave * 16 + jj + 8 * half;   // row within the 64-row tile
        int ig = i0 + il;
        int yq = ig & 31, xq = ig >> 5;
        S[jt][jj] += Pwl[il][yk - yq + 31] + Phl[il][xk - xq + 31];
      }
    }
    // ---- online softmax (rows live across 16-lane halves) ----
    float newm[8], fs[8], rs[8];
#pragma unroll
    for (int jj = 0; jj < 8; ++jj) {
      float mx = S[0][jj];
#pragma unroll
      for (int jt = 1; jt < 4; ++jt) mx = fmaxf(mx, S[jt][jj]);
#pragma unroll
      for (int off = 1; off < 16; off <<= 1) mx = fmaxf(mx, __shfl_xor(mx, off));
      newm[jj] = fmaxf(mrow[jj], mx);
      fs[jj] = __expf(mrow[jj] - newm[jj]);
      rs[jj] = 0.0f;
    }
#pragma unroll
    for (int jt = 0; jt < 4; ++jt)
#pragma unroll
      for (int jj = 0; jj < 8; ++jj) {
        float p = __expf(S[jt][jj] - newm[jj]);
        S[jt][jj] = p;
        rs[jj] += p;
      }
#pragma unroll
    for (int jj = 0; jj < 8; ++jj) {
      float s = rs[jj];
#pragma unroll
      for (int off = 1; off < 16; off <<= 1) s += __shfl_xor(s, off);
      lrow[jj] = lrow[jj] * fs[jj] + s;
      mrow[jj] = newm[jj];
    }
#pragma unroll
    for (int dt = 0; dt < 8; ++dt)
#pragma unroll
      for (int jj = 0; jj < 8; ++jj) O[dt][jj] *= fs[jj];

    // ---- P: C-layout -> A-layout via LDS, bf16 ----
#pragma unroll
    for (int jt = 0; jt < 4; ++jt)
#pragma unroll
      for (int jj = 0; jj < 8; ++jj)
        Pbuf[wave][jj + 8 * half][jt * 16 + nl] = f2bf(S[jt][jj]);
    __syncthreads();

    v16bf PA[2];
#pragma unroll
    for (int kc = 0; kc < 2; ++kc) {
      const u16* p = &Pbuf[wave][nl][kc * 32 + 16 * half];
      PA[kc] = ld_frag2(p, p + 8);
    }
    // ---- O += P.V  (V stored transposed: contiguous keys per d) ----
#pragma unroll
    for (int dt = 0; dt < 8; ++dt)
#pragma unroll
      for (int kc = 0; kc < 2; ++kc) {
        const u16* vp = Vt + ((size_t)bh * DH + dt * 16 + nl) * NN + j0 + kc * 32 + 16 * half;
        O[dt] = wmma_bf16(PA[kc], ld_fragB(vp), O[dt]);
      }
  }

  // ---- normalize and store: out[b, h*128+d, i] ----
  float inv[8];
#pragma unroll
  for (int jj = 0; jj < 8; ++jj) inv[jj] = 1.0f / lrow[jj];
  int b = bh >> 2, hh = bh & 3;
#pragma unroll
  for (int dt = 0; dt < 8; ++dt) {
    size_t base = ((size_t)(b * (NH * DH) + hh * DH + dt * 16 + nl)) * NN +
                  i0 + wave * 16 + 8 * half;
    float4 lo, hi;
    lo.x = O[dt][0] * inv[0]; lo.y = O[dt][1] * inv[1];
    lo.z = O[dt][2] * inv[2]; lo.w = O[dt][3] * inv[3];
    hi.x = O[dt][4] * inv[4]; hi.y = O[dt][5] * inv[5];
    hi.z = O[dt][6] * inv[6]; hi.w = O[dt][7] * inv[7];
    *(float4*)(out + base) = lo;
    *(float4*)(out + base + 4) = hi;
  }
}

// ---------------------------------------------------------------------------
extern "C" void kernel_launch(void* const* d_in, const int* in_sizes, int n_in,
                              void* d_out, int out_size, void* d_ws, size_t ws_size,
                              hipStream_t stream) {
  const float* x     = (const float*)d_in[0];
  const float* w_qkv = (const float*)d_in[1];
  const float* rel_h = (const float*)d_in[2];
  const float* rel_w = (const float*)d_in[3];
  float* out = (float*)d_out;

  char* p = (char*)d_ws;
  u16* wbf    = (u16*)p;                 p += (size_t)OC * CIN * 2;          // 1.5 MB
  u16* relwbf = (u16*)p;                 p += 64 * DH * 2;
  u16* relhbf = (u16*)p;                 p += 64 * DH * 2;
  u16* xT     = (u16*)p;                 p += (size_t)NB * NN * CIN * 2;     // 32 MB
  u16* Qb     = (u16*)p;                 p += (size_t)NB * NH * NN * DH * 2; // 32 MB
  u16* Kb     = (u16*)p;                 p += (size_t)NB * NH * NN * DH * 2; // 32 MB
  u16* Vt     = (u16*)p;                 p += (size_t)NB * NH * NN * DH * 2; // 32 MB
  float* Pw   = (float*)p;               p += (size_t)NB * NH * NN * 64 * 4; // 32 MB
  float* Ph   = (float*)p;               p += (size_t)NB * NH * NN * 64 * 4; // 32 MB

  k_cvt_w<<<dim3((OC * CIN) / 256), dim3(256), 0, stream>>>(w_qkv, wbf);
  k_cvt_rel<<<dim3(128), dim3(128), 0, stream>>>(rel_w, rel_h, relwbf, relhbf);
  k_transpose_x<<<dim3(NB * 8 * 16), dim3(256), 0, stream>>>(x, xT);
  k_qkv<<<dim3(NB * 64 * 24 / 4), dim3(128), 0, stream>>>(xT, wbf, Qb, Kb, Vt);
  k_relpos<<<dim3(NB * NH * 64 / 4), dim3(128), 0, stream>>>(Qb, relwbf, relhbf, Pw, Ph);
  k_attn<<<dim3(NB * NH * 16), dim3(128), 0, stream>>>(Qb, Kb, Vt, Pw, Ph, out);
}
